// HierarchicalTemporalAttention_85572928405570
// MI455X (gfx1250) — matrix operand order repaired and checked by reading, not compiled, not run on hardware
//
#include <hip/hip_runtime.h>
#include <hip/hip_bf16.h>

// Shapes from reference
#define T0   4096
#define B_   128
#define F_   128
#define H_   128
#define G_   384      // 3*HID
#define NSC  4

typedef __attribute__((ext_vector_type(16))) _Float16 v16h;
typedef __attribute__((ext_vector_type(8)))  _Float16 v8h;
typedef __attribute__((ext_vector_type(8)))  float    v8f;

static __device__ __forceinline__ v16h cat16(v8h lo, v8h hi) {
  return __builtin_shufflevector(lo, hi, 0,1,2,3,4,5,6,7,8,9,10,11,12,13,14,15);
}
static __device__ __forceinline__ v8f wmma_f16(v16h a, v16h b, v8f c) {
  // v_wmma_f32_16x16x32_f16  (neg_a, A, neg_b, B, c_mod, C, reuse_a, reuse_b)
  return __builtin_amdgcn_wmma_f32_16x16x32_f16(false, a, false, b, (short)0, c, false, false);
}
static __device__ __forceinline__ float fsig(float x)  { return 1.0f / (1.0f + __expf(-x)); }
static __device__ __forceinline__ float ftanh_(float x){ float e = __expf(-2.0f*x); return (1.0f - e)/(1.0f + e); }

// ---------------------------------------------------------------------------
// Pack a [128 x 384] f32 weight into WMMA B-fragment-major f16 layout:
// frag[(ct*4+ks)*32 + lane][e]  where  K = ks*32 + (lane>>4)*8 + (e&7) + (e>>3)*16,
//                                      N = ct*16 + (lane&15)
// ---------------------------------------------------------------------------
__global__ void pack_weight_kernel(const float* __restrict__ W, _Float16* __restrict__ out) {
  int idx = blockIdx.x * blockDim.x + threadIdx.x;   // 24*4*32*16 = 49152
  if (idx >= 24*4*32*16) return;
  int e    = idx & 15;
  int lane = (idx >> 4) & 31;
  int ks   = (idx >> 9) & 3;
  int ct   = idx >> 11;
  int K = ks*32 + ((lane >> 4) * 8) + (e & 7) + ((e >> 3) * 16);
  int N = ct*16 + (lane & 15);
  out[idx] = (_Float16)W[K * G_ + N];
}

// ---------------------------------------------------------------------------
// Temporal downsample (mean over `c` steps) + f32 -> f16, row-major [T_s*B, F]
// ---------------------------------------------------------------------------
__global__ void downsample_kernel(const float* __restrict__ x, _Float16* __restrict__ out,
                                  int c, int Tout) {
  int idx = blockIdx.x * blockDim.x + threadIdx.x;
  int total = Tout * B_ * F_;
  if (idx >= total) return;
  int bf = idx % (B_ * F_);
  int t  = idx / (B_ * F_);
  const float* p = x + (size_t)t * c * (B_ * F_) + bf;
  float acc = 0.f;
  for (int j = 0; j < c; ++j) acc += p[(size_t)j * (B_ * F_)];
  out[idx] = (_Float16)(acc * (1.0f / (float)c));
}

// ---------------------------------------------------------------------------
// Gi = X @ Wi + bi  via WMMA. Each wave loads A once (4 k-step fragments held
// in registers) and produces 6 output tiles (ct group) -> 6x less X traffic.
// Output stored PRE-SWIZZLED in the C-matrix lane layout:
//   Gi[(mt*24+ct)*256 + lane*8 + i]  (f16)
// ---------------------------------------------------------------------------
__launch_bounds__(256)
__global__ void gemm_gi_kernel(const _Float16* __restrict__ X,
                               const _Float16* __restrict__ wpack,
                               const float*    __restrict__ bi,
                               _Float16*       __restrict__ Gi,
                               int rowTiles) {
  int wave = threadIdx.x >> 5;
  int lane = threadIdx.x & 31;
  long task = (long)blockIdx.x * 8 + wave;
  int mt = (int)(task >> 2);                 // row tile
  int g  = (int)(task & 3);                  // ct group: 6 tiles each
  if (mt >= rowTiles) return;                // uniform per wave

  int n0  = lane & 15;
  int hi  = lane >> 4;
  int kb  = hi * 8;
  const _Float16* xrow = X + (size_t)(mt*16 + n0) * F_;

  v16h a[4];
#pragma unroll
  for (int ks = 0; ks < 4; ++ks)
    a[ks] = cat16(*(const v8h*)(xrow + ks*32 + kb),
                  *(const v8h*)(xrow + ks*32 + kb + 16));

#pragma unroll
  for (int i = 0; i < 6; ++i) {
    int ct = g*6 + i;
    float bv = bi[ct*16 + n0];
    v8f acc = { bv, bv, bv, bv, bv, bv, bv, bv };
#pragma unroll
    for (int ks = 0; ks < 4; ++ks) {
      const _Float16* bp = wpack + ((size_t)(ct*4 + ks)*32 + lane)*16;
      acc = wmma_f16(a[ks], cat16(*(const v8h*)bp, *(const v8h*)(bp + 8)), acc);
    }
    v8h st;
#pragma unroll
    for (int j = 0; j < 8; ++j) st[j] = (_Float16)acc[j];
    *(v8h*)(Gi + ((size_t)(mt*24 + ct)*32 + lane)*8) = st;
  }
}

// ---------------------------------------------------------------------------
// GRU recurrence. One block per (scale, 16-row batch tile): 32 blocks x 4 waves.
// Per step: wave q computes 6 of the 24 WMMA column tiles into an LDS
// accumulator exchange buffer, barrier, then applies the gate nonlinearity for
// its 2 hid-column tiles (h state in f32 registers, f16 row-major LDS mirror
// feeds next step's A fragments), barrier. gi for step t+1 is software-
// prefetched (global_prefetch_b8) to hide HBM latency on the serial path.
// ---------------------------------------------------------------------------
struct RecArgs {
  const _Float16* gi[NSC];     // pre-swizzled gi, tile-major
  const _Float16* wp[NSC];     // packed Wh
  const float*    bhn[NSC];
  float*          hT;          // [NSC][B_][H_]
};

__launch_bounds__(128)
__global__ void gru_recurrence_kernel(RecArgs args) {
  __shared__ __align__(16) _Float16 whp[24*4*32*16];   // 96 KB packed Wh
  __shared__ __align__(16) _Float16 hbuf[16][136];     // h mirror, 272B pitch
  __shared__ __align__(32) float    accf[24][32][8];   // 24 KB accum exchange

  int bx   = blockIdx.x;
  int s    = bx >> 3;                       // scale
  int w    = bx & 7;                        // batch row tile (16 rows)
  int tid  = threadIdx.x;
  int q    = tid >> 5;                      // wave 0..3
  int lane = tid & 31;
  int n0   = lane & 15;
  int hi   = lane >> 4;
  int kb   = hi * 8;
  int Tsteps = T0 >> (2 * s);               // 4096,1024,256,64

  const _Float16* gi  = args.gi[s];
  const float*    bhn = args.bhn[s];

  // cooperative copy of packed Wh into LDS
  {
    const v8h* src = (const v8h*)args.wp[s];
    v8h* dst = (v8h*)whp;
    for (int i = tid; i < (24*4*32*16)/8; i += 128) dst[i] = src[i];
  }
  for (int i = tid; i < 16*136; i += 128) (&hbuf[0][0])[i] = (_Float16)0.0f;

  float bh[2];
  v8f hreg[2];
#pragma unroll
  for (int u = 0; u < 2; ++u) {
    int jt = q*2 + u;
    bh[u] = bhn[jt*16 + n0];
    hreg[u] = (v8f)(0.0f);
  }
  __syncthreads();

  for (int t = 0; t < Tsteps; ++t) {
    // A fragments for this block's 16 batch rows (h at step t)
    v16h a[4];
#pragma unroll
    for (int ks = 0; ks < 4; ++ks) {
      const _Float16* hrow = &hbuf[n0][0];
      a[ks] = cat16(*(const v8h*)(hrow + ks*32 + kb),
                    *(const v8h*)(hrow + ks*32 + kb + 16));
    }
    const _Float16* gbase = gi + (size_t)(t*8 + w) * 24 * 256;
    // prefetch next step's gi tile group (12KB; 128 lanes x 128B lines)
    {
      int tn = (t + 1 < Tsteps) ? (t + 1) : t;
      const char* gnext = (const char*)(gi + (size_t)(tn*8 + w) * 24 * 256);
      __builtin_prefetch(gnext + (size_t)tid * 128, 0, 1);
    }
    // WMMA: this wave's 6 column tiles of gh = h @ Wh
#pragma unroll
    for (int i = 0; i < 6; ++i) {
      int ct = q*6 + i;
      v8f acc = (v8f)(0.0f);
#pragma unroll
      for (int ks = 0; ks < 4; ++ks) {
        const _Float16* bp = whp + ((ct*4 + ks)*32 + lane)*16;
        acc = wmma_f16(a[ks], cat16(*(const v8h*)bp, *(const v8h*)(bp + 8)), acc);
      }
      *(v8f*)&accf[ct][lane][0] = acc;
    }
    __syncthreads();
    // gate nonlinearity for this wave's 2 hid-column tiles
#pragma unroll
    for (int u = 0; u < 2; ++u) {
      int jt = q*2 + u;
      v8f ar = *(const v8f*)&accf[jt][lane][0];
      v8f az = *(const v8f*)&accf[8 + jt][lane][0];
      v8f an = *(const v8f*)&accf[16 + jt][lane][0];
      v8h gr = *(const v8h*)(gbase + (jt)*256      + lane*8);
      v8h gz = *(const v8h*)(gbase + (8 + jt)*256  + lane*8);
      v8h gn = *(const v8h*)(gbase + (16 + jt)*256 + lane*8);
      int col = jt*16 + n0;
#pragma unroll
      for (int e = 0; e < 8; ++e) {
        int m = e + hi*8;
        float r  = fsig((float)gr[e] + ar[e]);
        float z  = fsig((float)gz[e] + az[e]);
        float nn = ftanh_((float)gn[e] + r * (an[e] + bh[u]));
        float hn = (1.0f - z) * nn + z * hreg[u][e];
        hreg[u][e] = hn;
        hbuf[m][col] = (_Float16)hn;       // feeds next step's A fragments
      }
    }
    __syncthreads();
  }

  // write final hidden state (f32) for this scale / batch tile
  float* out = args.hT + (size_t)s * B_ * H_;
#pragma unroll
  for (int u = 0; u < 2; ++u) {
    int jt = q*2 + u;
    int col = jt*16 + n0;
#pragma unroll
    for (int e = 0; e < 8; ++e) {
      int b = w*16 + e + hi*8;
      out[(size_t)b * H_ + col] = hreg[u][e];
    }
  }
}

// ---------------------------------------------------------------------------
// q/k/v projections over ms = hT : [4,B,H] -> [4,B,HEADS*HEAD_DIM]
// ---------------------------------------------------------------------------
__global__ void qkv_kernel(const float* __restrict__ ms,
                           const float* __restrict__ Wq, const float* __restrict__ bq,
                           const float* __restrict__ Wk, const float* __restrict__ bk,
                           const float* __restrict__ Wv, const float* __restrict__ bv,
                           float* __restrict__ q, float* __restrict__ k, float* __restrict__ v) {
  int idx = blockIdx.x * blockDim.x + threadIdx.x;   // 4*128*128
  if (idx >= NSC * B_ * H_) return;
  int hd = idx & 127;
  int sb = idx >> 7;
  const float* m = ms + (size_t)sb * H_;
  float aq = bq[hd], ak = bk[hd], av = bv[hd];
  for (int f = 0; f < H_; ++f) {
    float mv = m[f];
    aq += mv * Wq[f*H_ + hd];
    ak += mv * Wk[f*H_ + hd];
    av += mv * Wv[f*H_ + hd];
  }
  q[idx] = aq * 0.25f;   // 1/sqrt(HEAD_DIM=16)
  k[idx] = ak;
  v[idx] = av;
}

// ---------------------------------------------------------------------------
// Per (s,head) softmax attention over length B_=128, HEAD_DIM=16.
// ---------------------------------------------------------------------------
__launch_bounds__(128)
__global__ void attn_kernel(const float* __restrict__ q, const float* __restrict__ k,
                            const float* __restrict__ v, float* __restrict__ o) {
  __shared__ float kt[B_][16];
  __shared__ float vt[B_][16];
  int sh = blockIdx.x;
  int s = sh >> 3, h = sh & 7;
  for (int i = threadIdx.x; i < B_*16; i += blockDim.x) {
    int bb = i >> 4, d = i & 15;
    size_t off = (((size_t)s*B_ + bb)*8 + h)*16 + d;
    kt[bb][d] = k[off];
    vt[bb][d] = v[off];
  }
  __syncthreads();
  int bq_ = threadIdx.x;
  float qr[16];
#pragma unroll
  for (int d = 0; d < 16; ++d) qr[d] = q[(((size_t)s*B_ + bq_)*8 + h)*16 + d];
  float mx = -1e30f;
  for (int bb = 0; bb < B_; ++bb) {
    float dot = 0.f;
#pragma unroll
    for (int d = 0; d < 16; ++d) dot += qr[d] * kt[bb][d];
    mx = fmaxf(mx, dot);
  }
  float sum = 0.f;
  float acc[16];
#pragma unroll
  for (int d = 0; d < 16; ++d) acc[d] = 0.f;
  for (int bb = 0; bb < B_; ++bb) {
    float dot = 0.f;
#pragma unroll
    for (int d = 0; d < 16; ++d) dot += qr[d] * kt[bb][d];
    float p = __expf(dot - mx);
    sum += p;
#pragma unroll
    for (int d = 0; d < 16; ++d) acc[d] += p * vt[bb][d];
  }
  float inv = 1.0f / sum;
#pragma unroll
  for (int d = 0; d < 16; ++d)
    o[(((size_t)s*B_ + bq_)*8 + h)*16 + d] = acc[d] * inv;
}

// pooled = mean_s(o) @ Wo + bo   : [B,H]
__global__ void pooled_kernel(const float* __restrict__ o, const float* __restrict__ Wo,
                              const float* __restrict__ bo, float* __restrict__ pooled) {
  int idx = blockIdx.x * blockDim.x + threadIdx.x;   // 128*128
  if (idx >= B_ * H_) return;
  int f = idx & 127, b = idx >> 7;
  float acc = bo[f];
  for (int hd = 0; hd < H_; ++hd) {
    float om = 0.25f * (o[((size_t)0*B_ + b)*H_ + hd] + o[((size_t)1*B_ + b)*H_ + hd] +
                        o[((size_t)2*B_ + b)*H_ + hd] + o[((size_t)3*B_ + b)*H_ + hd]);
    acc += om * Wo[hd*H_ + f];
  }
  pooled[idx] = acc;
}

// out = pooled @ Wd + bd
__global__ void final_kernel(const float* __restrict__ pooled, const float* __restrict__ Wd,
                             const float* __restrict__ bd, float* __restrict__ out) {
  int idx = blockIdx.x * blockDim.x + threadIdx.x;   // 128*128
  if (idx >= B_ * H_) return;
  int f2 = idx & 127, b = idx >> 7;
  float acc = bd[f2];
  for (int f = 0; f < H_; ++f) acc += pooled[(size_t)b*H_ + f] * Wd[f*H_ + f2];
  out[idx] = acc;
}

// ---------------------------------------------------------------------------
extern "C" void kernel_launch(void* const* d_in, const int* in_sizes, int n_in,
                              void* d_out, int out_size, void* d_ws, size_t ws_size,
                              hipStream_t stream) {
  (void)in_sizes; (void)n_in; (void)out_size; (void)ws_size;
  const float* x = (const float*)d_in[0];
  const float *Wi[NSC], *Wh[NSC], *bi[NSC], *bhn[NSC];
  for (int s = 0; s < NSC; ++s) {
    Wi[s]  = (const float*)d_in[1 + 4*s];
    Wh[s]  = (const float*)d_in[2 + 4*s];
    bi[s]  = (const float*)d_in[3 + 4*s];
    bhn[s] = (const float*)d_in[4 + 4*s];
  }
  const float* Wq = (const float*)d_in[17]; const float* bq = (const float*)d_in[18];
  const float* Wk = (const float*)d_in[19]; const float* bk = (const float*)d_in[20];
  const float* Wv = (const float*)d_in[21]; const float* bv = (const float*)d_in[22];
  const float* Wo = (const float*)d_in[23]; const float* bo = (const float*)d_in[24];
  const float* Wd = (const float*)d_in[25]; const float* bd = (const float*)d_in[26];

  // workspace carve-up
  char* ws = (char*)d_ws;
  size_t off = 0;
  auto alloc = [&](size_t bytes) -> void* {
    off = (off + 255) & ~(size_t)255;
    void* p = ws + off;
    off += bytes;
    return p;
  };

  int Ts[NSC];
  _Float16 *X16[NSC], *WiP[NSC], *WhP[NSC], *Gi[NSC];
  for (int s = 0; s < NSC; ++s) {
    Ts[s] = T0 >> (2*s);
    X16[s] = (_Float16*)alloc((size_t)Ts[s] * B_ * F_ * sizeof(_Float16));
  }
  for (int s = 0; s < NSC; ++s) WiP[s] = (_Float16*)alloc((size_t)24*4*32*16 * sizeof(_Float16));
  for (int s = 0; s < NSC; ++s) WhP[s] = (_Float16*)alloc((size_t)24*4*32*16 * sizeof(_Float16));
  for (int s = 0; s < NSC; ++s) Gi[s]  = (_Float16*)alloc((size_t)Ts[s] * B_ * G_ * sizeof(_Float16));
  float* hT     = (float*)alloc((size_t)NSC * B_ * H_ * sizeof(float));
  float* qb     = (float*)alloc((size_t)NSC * B_ * H_ * sizeof(float));
  float* kb_    = (float*)alloc((size_t)NSC * B_ * H_ * sizeof(float));
  float* vb     = (float*)alloc((size_t)NSC * B_ * H_ * sizeof(float));
  float* ob     = (float*)alloc((size_t)NSC * B_ * H_ * sizeof(float));
  float* pooled = (float*)alloc((size_t)B_ * H_ * sizeof(float));

  // 1) pack weights into WMMA B-fragment layout
  for (int s = 0; s < NSC; ++s) {
    pack_weight_kernel<<<192, 256, 0, stream>>>(Wi[s], WiP[s]);
    pack_weight_kernel<<<192, 256, 0, stream>>>(Wh[s], WhP[s]);
  }
  // 2) downsample + f16 convert
  for (int s = 0; s < NSC; ++s) {
    int c = 1 << (2*s);
    int total = Ts[s] * B_ * F_;
    downsample_kernel<<<(total + 255)/256, 256, 0, stream>>>(x, X16[s], c, Ts[s]);
  }
  // 3) big parallel WMMA GEMMs: gi = X @ Wi + bi (output pre-swizzled)
  for (int s = 0; s < NSC; ++s) {
    int rowTiles = Ts[s] * (B_/16);            // Ts*8
    int blocks   = Ts[s] * 4;                  // rowTiles*4 tasks / 8 waves
    gemm_gi_kernel<<<blocks, 256, 0, stream>>>(X16[s], WiP[s], bi[s], Gi[s], rowTiles);
  }
  // 4) sequential GRU recurrences: 32 blocks (scale x batch tile), 4 waves each
  RecArgs ra;
  for (int s = 0; s < NSC; ++s) { ra.gi[s] = Gi[s]; ra.wp[s] = WhP[s]; ra.bhn[s] = bhn[s]; }
  ra.hT = hT;
  gru_recurrence_kernel<<<NSC*8, 128, 0, stream>>>(ra);
  // 5) tiny attention head + dense layers
  qkv_kernel<<<(NSC*B_*H_ + 255)/256, 256, 0, stream>>>(hT, Wq, bq, Wk, bk, Wv, bv, qb, kb_, vb);
  attn_kernel<<<NSC*8, 128, 0, stream>>>(qb, kb_, vb, ob);
  pooled_kernel<<<(B_*H_ + 255)/256, 256, 0, stream>>>(ob, Wo, bo, pooled);
  final_kernel<<<(B_*H_ + 255)/256, 256, 0, stream>>>(pooled, Wd, bd, (float*)d_out);
}
